// TransformerLayer_63522566308290
// MI455X (gfx1250) — matrix operand order repaired and checked by reading, not compiled, not run on hardware
//
#include <hip/hip_runtime.h>
#include <hip/hip_bf16.h>
#include <math.h>

// ---------------------------------------------------------------------------
// CDNA5 (gfx1250) graph-transformer layer.
// GEMMs run through v_wmma_f32_16x16x32_bf16 (fp32 accumulate).
// Scatter/softmax phases use fp32 atomics that stay resident in the 192MB L2.
// ---------------------------------------------------------------------------

typedef __bf16 bf16_t;
typedef __attribute__((ext_vector_type(16))) __bf16 bf16x16;
typedef __attribute__((ext_vector_type(8)))  float  f32x8;
typedef __attribute__((ext_vector_type(4)))  int    i32x4;

#define HID   128
#define HC    512        // H*C
#define NHEAD 4
#define KPX   160        // IN_DIM 137 padded to 5*32
#define KPE   64         // EDIM 60 padded to 2*32
#define TWO_PI 6.28318530717958647692f

// ------------------------------- helpers -----------------------------------

__device__ __forceinline__ void atomicMaxF(float* addr, float val) {
    // classic float-max via int/uint atomics (works for mixed signs)
    if (val >= 0.f) atomicMax((int*)addr, __float_as_int(val));
    else            atomicMin((unsigned int*)addr, __float_as_uint(val));
}

// ------------------------- init scratch buffers ----------------------------

__global__ void init_ws(float* mbuf, float* sbuf, float* agg, int N) {
    int tid = blockIdx.x * blockDim.x + threadIdx.x;
    if (tid < N * NHEAD) { mbuf[tid] = -__builtin_huge_valf(); sbuf[tid] = 0.f; }
    if (tid < N * HC)    agg[tid] = 0.f;
}

// -------------------- LayerNorm + concat -> x_bf16 [N,160] -----------------

__global__ void node_prep(const float* __restrict__ nf, const float* __restrict__ frac,
                          const float* __restrict__ lat, const int* __restrict__ n2g,
                          const float* __restrict__ gamma, const float* __restrict__ beta,
                          bf16_t* __restrict__ xbf, int N)
{
    const int wave = threadIdx.x >> 5;
    const int lane = threadIdx.x & 31;
    const int n = blockIdx.x * (blockDim.x >> 5) + wave;
    if (n >= N) return;

    const float* row = nf + (size_t)n * HID;
    float v0 = row[lane], v1 = row[lane + 32], v2 = row[lane + 64], v3 = row[lane + 96];
    float s  = v0 + v1 + v2 + v3;
    float ss = v0*v0 + v1*v1 + v2*v2 + v3*v3;
    #pragma unroll
    for (int m = 16; m >= 1; m >>= 1) {
        s  += __shfl_xor(s,  m, 32);
        ss += __shfl_xor(ss, m, 32);
    }
    float mu  = s * (1.f / HID);
    float var = ss * (1.f / HID) - mu * mu;      // biased variance (torch default)
    float inv = rsqrtf(var + 1e-5f);

    bf16_t* xr = xbf + (size_t)n * KPX;
    xr[lane]      = (bf16_t)((v0 - mu) * inv * gamma[lane]      + beta[lane]);
    xr[lane + 32] = (bf16_t)((v1 - mu) * inv * gamma[lane + 32] + beta[lane + 32]);
    xr[lane + 64] = (bf16_t)((v2 - mu) * inv * gamma[lane + 64] + beta[lane + 64]);
    xr[lane + 96] = (bf16_t)((v3 - mu) * inv * gamma[lane + 96] + beta[lane + 96]);

    // tail: 6 lattice feats, 3 frac coords, zero pad to 160
    int g = n2g[n];
    float tail;
    if      (lane < 6) tail = lat[(size_t)g * 6 + lane];
    else if (lane < 9) tail = frac[(size_t)n * 3 + (lane - 6)];
    else               tail = 0.f;
    xr[HID + lane] = (bf16_t)tail;               // covers 128..159
}

// ---------------- sinusoid edge embedding -> bf16 [E,64] -------------------

__global__ void edge_emb_kernel(const float* __restrict__ fd, bf16_t* __restrict__ ebf, int E) {
    int tid = blockIdx.x * blockDim.x + threadIdx.x;
    if (tid >= E * KPE) return;
    int e = tid >> 6, j = tid & 63;
    float val = 0.f;
    if (j < 60) {
        int jj = (j < 30) ? j : j - 30;
        int d = jj / 10, f = jj % 10;
        float ang = fd[(size_t)e * 3 + d] * (TWO_PI * (float)f);
        val = (j < 30) ? sinf(ang) : cosf(ang);
    }
    ebf[(size_t)e * KPE + j] = (bf16_t)val;
}

// --------- pre-pack W[K][Ncols] into WMMA B-operand register image ---------
// layout: Wp[((ct*nkb + kb)*32 + lane)*16 + i]  where
//   n = ct*16 + lane%16 ; k = kb*32 + (lane/16)*16 + i   (zero-padded K)

__global__ void pack_w(const float* __restrict__ W, bf16_t* __restrict__ Wp,
                       int Kreal, int Kp, int Ncols)
{
    int tid = blockIdx.x * blockDim.x + threadIdx.x;
    int nkb = Kp >> 5;
    int total = (Ncols >> 4) * nkb * 32 * 16;
    if (tid >= total) return;
    int i    = tid & 15;
    int lane = (tid >> 4) & 31;
    int rest = tid >> 9;
    int kb   = rest % nkb;
    int ct   = rest / nkb;
    int n = ct * 16 + (lane & 15);
    int k = kb * 32 + (lane >> 4) * 16 + i;
    float v = (k < Kreal) ? W[(size_t)k * Ncols + n] : 0.f;
    Wp[tid] = (bf16_t)v;
}

// ------------------------- WMMA bf16 GEMM ----------------------------------
// C[M x Ncols] = A_bf16[M x Kp(row-major,lda)] @ Bpacked + bias
// block = 128 threads (4 waves); wave -> one 16x16 output tile.

template<bool OUT_BF16>
__global__ void gemm_wmma(const bf16_t* __restrict__ A, int lda,
                          const bf16_t* __restrict__ Bp,
                          const float* __restrict__ bias,
                          void* __restrict__ Cout, int ldc,
                          int M, int Kp, int Ncols)
{
    const int lane = threadIdx.x & 31;
    const int wave = threadIdx.x >> 5;
    const int rowTile = blockIdx.x;
    const int colTile = blockIdx.y * 4 + wave;
    if (colTile * 16 >= Ncols) return;
    const int half = lane >> 4;
    const int lmod = lane & 15;

    int row  = rowTile * 16 + lmod;
    int rowc = (row < M) ? row : (M - 1);          // clamp so OOB rows load safely

    const int nkb = Kp >> 5;
    f32x8 acc = {};
    union { bf16x16 v; i32x4 q[2]; } a, b;
    const bf16_t* Arow = A  + (size_t)rowc * lda;
    const bf16_t* Bw   = Bp + ((size_t)colTile * nkb) * 512 + (size_t)lane * 16;

    for (int kb = 0; kb < nkb; ++kb) {
        // A fragment: lane = row, contiguous 16 K per half-wave (matches packed B)
        const bf16_t* ap = Arow + kb * 32 + half * 16;
        a.q[0] = *(const i32x4*)(ap);              // k_local = half*16 .. +7
        a.q[1] = *(const i32x4*)(ap + 8);          // k_local = half*16+8 .. +15
        // B fragment: pre-packed register image, one contiguous 32B load
        const bf16_t* bp = Bw + (size_t)kb * 512;
        b.q[0] = *(const i32x4*)(bp);
        b.q[1] = *(const i32x4*)(bp + 8);
        acc = __builtin_amdgcn_wmma_f32_16x16x32_bf16(
                  false, a.v, false, b.v, (short)0, acc, false, false);
    }

    const int n = colTile * 16 + lmod;
    const float bval = bias ? bias[n] : 0.f;
    #pragma unroll
    for (int r = 0; r < 8; ++r) {
        int m = rowTile * 16 + half * 8 + r;       // C/D layout: row = r + 8*half
        if (m < M) {
            float val = acc[r] + bval;
            if (OUT_BF16) ((bf16_t*)Cout)[(size_t)m * ldc + n] = (bf16_t)val;
            else          ((float*) Cout)[(size_t)m * ldc + n] = val;
        }
    }
}

// ------------------ attention scores + segment max -------------------------
// one wave per edge; lane covers 16 contiguous channels of the 512-vector,
// 8 lanes per head -> shfl_xor reduce within groups of 8.

__global__ void score_kernel(const int* __restrict__ ei,
                             const float* __restrict__ q, const float* __restrict__ k,
                             const bf16_t* __restrict__ ebf,
                             float* __restrict__ alpha, float* __restrict__ mbuf, int E)
{
    const int wave = threadIdx.x >> 5, lane = threadIdx.x & 31;
    const int e = blockIdx.x * (blockDim.x >> 5) + wave;
    if (e >= E) return;
    const int src = ei[e], dst = ei[E + e];
    const float*  qp = q   + (size_t)dst * HC;
    const float*  kp = k   + (size_t)src * HC;
    const bf16_t* ep = ebf + (size_t)e   * HC;
    const int base = lane * 16;
    float sum = 0.f;
    #pragma unroll
    for (int t = 0; t < 16; ++t) {
        int idx = base + t;
        sum += qp[idx] * (kp[idx] + (float)ep[idx]);
    }
    sum += __shfl_xor(sum, 1, 32);
    sum += __shfl_xor(sum, 2, 32);
    sum += __shfl_xor(sum, 4, 32);
    if ((lane & 7) == 0) {
        int head = lane >> 3;
        float sc = sum * 0.08838834764831845f;     // 1/sqrt(128)
        alpha[(size_t)e * NHEAD + head] = sc;
        atomicMaxF(mbuf + (size_t)dst * NHEAD + head, sc);
    }
}

// ------------------------ exp + segment sum --------------------------------

__global__ void expsum_kernel(const int* __restrict__ ei, float* __restrict__ alpha,
                              const float* __restrict__ mbuf, float* __restrict__ sbuf, int E)
{
    int tid = blockIdx.x * blockDim.x + threadIdx.x;
    if (tid >= E * NHEAD) return;
    int e = tid >> 2, h = tid & 3;
    int dst = ei[E + e];
    float ex = expf(alpha[tid] - mbuf[(size_t)dst * NHEAD + h]);
    alpha[tid] = ex;
    atomicAdd(sbuf + (size_t)dst * NHEAD + h, ex);
}

// ------------------- weighted message scatter-add --------------------------
// agg[N,512] is 20MB -> L2 resident, so these atomics never touch HBM hot.

__global__ void agg_kernel(const int* __restrict__ ei, const float* __restrict__ v,
                           const bf16_t* __restrict__ ebf,
                           const float* __restrict__ alpha, const float* __restrict__ sbuf,
                           float* __restrict__ agg, int E)
{
    const int wave = threadIdx.x >> 5, lane = threadIdx.x & 31;
    const int e = blockIdx.x * (blockDim.x >> 5) + wave;
    if (e >= E) return;
    const int src = ei[e], dst = ei[E + e];
    const int head = lane >> 3;
    const float w = alpha[(size_t)e * NHEAD + head] /
                    (sbuf[(size_t)dst * NHEAD + head] + 1e-16f);
    const float*  vp = v   + (size_t)src * HC;
    const bf16_t* ep = ebf + (size_t)e   * HC;
    float* ag = agg + (size_t)dst * HC;
    const int base = lane * 16;
    #pragma unroll
    for (int t = 0; t < 16; ++t) {
        int idx = base + t;
        atomicAdd(ag + idx, (vp[idx] + (float)ep[idx]) * w);
    }
}

// ---------------- head mean + skip + SiLU + residual -----------------------

__global__ void finalize_kernel(const float* __restrict__ nf, const float* __restrict__ agg,
                                const float* __restrict__ skip, float* __restrict__ out, int N)
{
    const int wave = threadIdx.x >> 5, lane = threadIdx.x & 31;
    const int n = blockIdx.x * (blockDim.x >> 5) + wave;
    if (n >= N) return;
    const float* ag = agg + (size_t)n * HC;
    #pragma unroll
    for (int t = 0; t < 4; ++t) {
        int c = lane + t * 32;
        float o = 0.25f * (ag[c] + ag[128 + c] + ag[256 + c] + ag[384 + c])
                + skip[(size_t)n * HID + c];
        float y = o / (1.f + expf(-o));            // silu
        out[(size_t)n * HID + c] = nf[(size_t)n * HID + c] + y;
    }
}

// ------------------------------- launch ------------------------------------

extern "C" void kernel_launch(void* const* d_in, const int* in_sizes, int n_in,
                              void* d_out, int out_size, void* d_ws, size_t ws_size,
                              hipStream_t stream)
{
    const float* nf    = (const float*)d_in[0];
    const float* frac  = (const float*)d_in[1];
    const float* lat   = (const float*)d_in[2];
    const int*   ei    = (const int*)  d_in[3];
    const int*   n2g   = (const int*)  d_in[4];
    const float* fdiff = (const float*)d_in[5];
    const float* gamma = (const float*)d_in[6];
    const float* beta  = (const float*)d_in[7];
    const float* Wq = (const float*)d_in[8];   const float* bq = (const float*)d_in[9];
    const float* Wk = (const float*)d_in[10];  const float* bk = (const float*)d_in[11];
    const float* Wv = (const float*)d_in[12];  const float* bv = (const float*)d_in[13];
    const float* We = (const float*)d_in[14];
    const float* Wsk = (const float*)d_in[15]; const float* bsk = (const float*)d_in[16];

    const int N = in_sizes[0] / HID;
    const int E = in_sizes[3] / 2;

    // ---- carve workspace (256B aligned regions) ----
    char* wsb = (char*)d_ws;
    size_t off = 0;
    auto carve = [&](size_t bytes) -> char* {
        char* p = wsb + off;
        off = (off + bytes + 255) & ~(size_t)255;
        return p;
    };
    bf16_t* xbf   = (bf16_t*)carve((size_t)N * KPX * 2);
    bf16_t* embf  = (bf16_t*)carve((size_t)E * KPE * 2);
    bf16_t* WqP   = (bf16_t*)carve((size_t)KPX * HC * 2);
    bf16_t* WkP   = (bf16_t*)carve((size_t)KPX * HC * 2);
    bf16_t* WvP   = (bf16_t*)carve((size_t)KPX * HC * 2);
    bf16_t* WsP   = (bf16_t*)carve((size_t)KPX * HID * 2);
    bf16_t* WeP   = (bf16_t*)carve((size_t)KPE * HC * 2);
    float*  qv    = (float*) carve((size_t)N * HC * 4);
    float*  kv    = (float*) carve((size_t)N * HC * 4);
    float*  vv    = (float*) carve((size_t)N * HC * 4);
    float*  skip  = (float*) carve((size_t)N * HID * 4);
    bf16_t* ebf   = (bf16_t*)carve((size_t)E * HC * 2);
    float*  alpha = (float*) carve((size_t)E * NHEAD * 4);
    float*  mbuf  = (float*) carve((size_t)N * NHEAD * 4);
    float*  sbuf  = (float*) carve((size_t)N * NHEAD * 4);
    float*  agg   = (float*) carve((size_t)N * HC * 4);

    auto cdiv = [](int a, int b) { return (a + b - 1) / b; };

    // 1) init softmax stats + aggregation buffer
    init_ws<<<cdiv(N * HC, 256), 256, 0, stream>>>(mbuf, sbuf, agg, N);

    // 2) LayerNorm + concat (one wave per node)
    node_prep<<<cdiv(N, 4), 128, 0, stream>>>(nf, frac, lat, n2g, gamma, beta, xbf, N);

    // 3) sinusoid edge embedding
    edge_emb_kernel<<<cdiv(E * KPE, 256), 256, 0, stream>>>(fdiff, embf, E);

    // 4) pre-pack weights into WMMA B-operand register images
    pack_w<<<cdiv((HC  / 16) * (KPX / 32) * 512, 256), 256, 0, stream>>>(Wq,  WqP, 137, KPX, HC);
    pack_w<<<cdiv((HC  / 16) * (KPX / 32) * 512, 256), 256, 0, stream>>>(Wk,  WkP, 137, KPX, HC);
    pack_w<<<cdiv((HC  / 16) * (KPX / 32) * 512, 256), 256, 0, stream>>>(Wv,  WvP, 137, KPX, HC);
    pack_w<<<cdiv((HID / 16) * (KPX / 32) * 512, 256), 256, 0, stream>>>(Wsk, WsP, 137, KPX, HID);
    pack_w<<<cdiv((HC  / 16) * (KPE / 32) * 512, 256), 256, 0, stream>>>(We,  WeP,  60, KPE, HC);

    // 5) WMMA GEMMs
    dim3 gqkv(cdiv(N, 16), HC / 64);
    gemm_wmma<false><<<gqkv, 128, 0, stream>>>(xbf, KPX, WqP, bq, qv,   HC,  N, KPX, HC);
    gemm_wmma<false><<<gqkv, 128, 0, stream>>>(xbf, KPX, WkP, bk, kv,   HC,  N, KPX, HC);
    gemm_wmma<false><<<gqkv, 128, 0, stream>>>(xbf, KPX, WvP, bv, vv,   HC,  N, KPX, HC);
    dim3 gsk(cdiv(N, 16), HID / 64);
    gemm_wmma<false><<<gsk,  128, 0, stream>>>(xbf, KPX, WsP, bsk, skip, HID, N, KPX, HID);
    dim3 ge(cdiv(E, 16), HC / 64);
    gemm_wmma<true><<<ge,    128, 0, stream>>>(embf, KPE, WeP, nullptr, ebf, HC, E, KPE, HC);

    // 6) attention scores + segment max (one wave per edge)
    score_kernel<<<cdiv(E, 8), 256, 0, stream>>>(ei, qv, kv, ebf, alpha, mbuf, E);

    // 7) exp + segment sum
    expsum_kernel<<<cdiv(E * NHEAD, 256), 256, 0, stream>>>(ei, alpha, mbuf, sbuf, E);

    // 8) weighted message scatter-add (one wave per edge)
    agg_kernel<<<cdiv(E, 8), 256, 0, stream>>>(ei, vv, ebf, alpha, sbuf, agg, E);

    // 9) head mean + skip + SiLU + residual
    finalize_kernel<<<cdiv(N, 4), 128, 0, stream>>>(nf, agg, skip, (float*)d_out, N);
}
